// ResidualBlock_basic_12137577579238
// MI455X (gfx1250) — matrix operand order repaired and checked by reading, not compiled, not run on hardware
//
#include <hip/hip_runtime.h>

#define EPS_BN 1e-4f
#define NPTS 150000
#define CCH  64
#define KOFF 27
#define PPAIR 50000
#define PTILES (PPAIR / 16)          // 3125, exact
#define NTILES (KOFF * PTILES)       // 84375 wave-tiles
#define WAVES_PER_BLK 8

typedef __attribute__((ext_vector_type(16))) _Float16 v16h;
typedef __attribute__((ext_vector_type(8)))  _Float16 v8h;
typedef __attribute__((ext_vector_type(8)))  float    v8f;

// ---------------- elementwise kernels ----------------

// eval-mode BatchNorm + ReLU, f32 in -> f16 out (WMMA A-operand precision)
__global__ void bnrelu_f32_to_f16(const float* __restrict__ x,
                                  const float* __restrict__ gamma,
                                  const float* __restrict__ beta,
                                  const float* __restrict__ mean,
                                  const float* __restrict__ var,
                                  _Float16* __restrict__ y, int total) {
  int i = blockIdx.x * blockDim.x + threadIdx.x;
  if (i >= total) return;
  int c = i & (CCH - 1);
  float s = gamma[c] * rsqrtf(var[c] + EPS_BN);
  float v = (x[i] - mean[c]) * s + beta[c];
  y[i] = (_Float16)(v > 0.0f ? v : 0.0f);
}

// W[k][cin][cout] f32  ->  Wt[k][cout][cin] f16 (row-contiguous B fragments)
__global__ void prep_weights(const float* __restrict__ W, _Float16* __restrict__ Wt) {
  int i = blockIdx.x * blockDim.x + threadIdx.x;
  if (i >= KOFF * CCH * CCH) return;
  int k = i / (CCH * CCH);
  int r = i - k * (CCH * CCH);
  int cin = r >> 6, cout = r & 63;
  Wt[((size_t)k * CCH + cout) * CCH + cin] = (_Float16)W[i];
}

__global__ void zero_f32(float* __restrict__ p, int n) {
  int i = blockIdx.x * blockDim.x + threadIdx.x;
  if (i < n) p[i] = 0.0f;
}

__global__ void copy_f32(const float* __restrict__ s, float* __restrict__ d, int n) {
  int i = blockIdx.x * blockDim.x + threadIdx.x;
  if (i < n) d[i] = s[i];
}

// ---------------- sparse-conv WMMA kernel ----------------
// One wave32 per (offset k, 16-row tile of P). Gather rows of y (f16) per
// rulebook, 16x16x32 f16 WMMA against pre-transposed weights, f32 atomic
// scatter-add into out per rb_out.
__global__ void __launch_bounds__(WAVES_PER_BLK * 32)
sconv_wmma(const _Float16* __restrict__ y,    // [NPTS, 64] f16
           const _Float16* __restrict__ Wt,   // [27, 64cout, 64cin] f16
           const int* __restrict__ rb_in,     // [27, P]
           const int* __restrict__ rb_out,    // [27, P]
           float* __restrict__ out,           // [NPTS, 64] f32 (accumulated)
           int ntiles) {
  const int lane = threadIdx.x & 31;
  const int wv   = threadIdx.x >> 5;
  int wt = blockIdx.x * WAVES_PER_BLK + wv;
  const bool valid = (wt < ntiles);
  if (!valid) wt = ntiles - 1;            // clamp: keep EXEC full through WMMA

  const int k = wt / PTILES;
  const int t = wt - k * PTILES;
  const int pbase = k * PPAIR + t * 16;

  const int g   = lane >> 4;              // half-wave group (ISA A/B/C layouts)
  const int col = lane & 15;

  int in_idx = 0, out_idx = 0;
  if (lane < 16) {
    in_idx  = rb_in [pbase + lane];
    out_idx = rb_out[pbase + lane];
  }
  // broadcast this lane's A-row index (row M = lane&15 for both half-waves)
  const int rowA = __shfl(in_idx, col, 32);
  const _Float16* __restrict__ rowp  = y + (size_t)rowA * CCH;
  const _Float16* __restrict__ wbase = Wt + (size_t)k * CCH * CCH;

  v8f acc0 = {}, acc1 = {}, acc2 = {}, acc3 = {};

#pragma unroll
  for (int c = 0; c < 2; ++c) {           // two K=32 chunks over Cin=64
    // A 16x32 f16 fragment: lane holds K = c*32 + 16g + j (contiguous)
    union { v16h v; v8h h[2]; } au;
    au.h[0] = *(const v8h*)(rowp + c * 32 +     16 * g);
    au.h[1] = *(const v8h*)(rowp + c * 32 + 8 + 16 * g);
    const v16h a = au.v;

    // B 32x16 f16 fragments: lane N=col holds K = c*32 + 16g + j (contiguous)
    const _Float16* __restrict__ wp = wbase + c * 32 + 16 * g;
    const v16h b0 = *(const v16h*)(wp + (size_t)(col     ) * CCH);
    const v16h b1 = *(const v16h*)(wp + (size_t)(col + 16) * CCH);
    const v16h b2 = *(const v16h*)(wp + (size_t)(col + 32) * CCH);
    const v16h b3 = *(const v16h*)(wp + (size_t)(col + 48) * CCH);

    acc0 = __builtin_amdgcn_wmma_f32_16x16x32_f16(false, a, false, b0, (short)0, acc0, false, false);
    acc1 = __builtin_amdgcn_wmma_f32_16x16x32_f16(false, a, false, b1, (short)0, acc1, false, false);
    acc2 = __builtin_amdgcn_wmma_f32_16x16x32_f16(false, a, false, b2, (short)0, acc2, false, false);
    acc3 = __builtin_amdgcn_wmma_f32_16x16x32_f16(false, a, false, b3, (short)0, acc3, false, false);
  }

  if (valid) {                            // wave-uniform branch; EXEC stays full
#pragma unroll
    for (int r = 0; r < 8; ++r) {         // D row M = r + 8g in VGPR r
      const int orow = __shfl(out_idx, r + 8 * g, 32);
      float* op = out + (size_t)orow * CCH + col;
      atomicAdd(op +  0, acc0[r]);
      atomicAdd(op + 16, acc1[r]);
      atomicAdd(op + 32, acc2[r]);
      atomicAdd(op + 48, acc3[r]);
    }
  }
}

// ---------------- host orchestration ----------------

extern "C" void kernel_launch(void* const* d_in, const int* in_sizes, int n_in,
                              void* d_out, int out_size, void* d_ws, size_t ws_size,
                              hipStream_t stream) {
  const float* feats = (const float*)d_in[0];
  const float* g1 = (const float*)d_in[1];
  const float* b1 = (const float*)d_in[2];
  const float* m1 = (const float*)d_in[3];
  const float* v1 = (const float*)d_in[4];
  const float* W1 = (const float*)d_in[5];
  const float* g2 = (const float*)d_in[6];
  const float* b2 = (const float*)d_in[7];
  const float* m2 = (const float*)d_in[8];
  const float* v2 = (const float*)d_in[9];
  const float* W2 = (const float*)d_in[10];
  const int* rb_in  = (const int*)d_in[11];
  const int* rb_out = (const int*)d_in[12];
  float* out = (float*)d_out;

  // workspace layout (all 16B-aligned offsets)
  char* ws = (char*)d_ws;
  const size_t wbytes = (size_t)KOFF * CCH * CCH * sizeof(_Float16); // 221184
  _Float16* Wt1 = (_Float16*)(ws);
  _Float16* Wt2 = (_Float16*)(ws + wbytes);
  _Float16* yh  = (_Float16*)(ws + 2 * wbytes);                      // N*64 f16
  float*    mid = (float*)   (ws + 2 * wbytes + (size_t)NPTS * CCH * sizeof(_Float16));

  const int total = NPTS * CCH;
  const int wtot  = KOFF * CCH * CCH;
  const int eb = 256;

  // 1) weight precompute (f32 -> f16, transposed to [k][cout][cin])
  prep_weights<<<(wtot + eb - 1) / eb, eb, 0, stream>>>(W1, Wt1);
  prep_weights<<<(wtot + eb - 1) / eb, eb, 0, stream>>>(W2, Wt2);

  // 2) y1 = BNReLU1(feats) in f16
  bnrelu_f32_to_f16<<<(total + eb - 1) / eb, eb, 0, stream>>>(feats, g1, b1, m1, v1, yh, total);

  // 3) mid = 0 ; mid += conv1(y1)
  zero_f32<<<(total + eb - 1) / eb, eb, 0, stream>>>(mid, total);
  const int cblocks = (NTILES + WAVES_PER_BLK - 1) / WAVES_PER_BLK;
  sconv_wmma<<<cblocks, WAVES_PER_BLK * 32, 0, stream>>>(yh, Wt1, rb_in, rb_out, mid, NTILES);

  // 4) y2 = BNReLU2(mid) in f16 (reuse yh)
  bnrelu_f32_to_f16<<<(total + eb - 1) / eb, eb, 0, stream>>>(mid, g2, b2, m2, v2, yh, total);

  // 5) out = feats ; out += conv2(y2)   (residual fused into scatter base)
  copy_f32<<<(total + eb - 1) / eb, eb, 0, stream>>>(feats, out, total);
  sconv_wmma<<<cblocks, WAVES_PER_BLK * 32, 0, stream>>>(yh, Wt2, rb_in, rb_out, out, NTILES);
}
